// GIN_81930796138764
// MI455X (gfx1250) — compile-verified, matrix operand order checked by reading, not run on hardware
//
#include <hip/hip_runtime.h>
#include <hip/hip_bf16.h>

typedef __attribute__((ext_vector_type(16))) _Float16 v16h;
typedef __attribute__((ext_vector_type(8)))  float    v8f;

#define NN 50000
#define EE 800000
#define GG 256
#define BN_EPS 1e-5f

// ---------------------------------------------------------------------------
// Elementwise helpers
// ---------------------------------------------------------------------------
__global__ __launch_bounds__(256) void copy4_kernel(const float* __restrict__ src,
                                                    float* __restrict__ dst, int n4) {
    int t = blockIdx.x * blockDim.x + threadIdx.x;
    if (t < n4) ((float4*)dst)[t] = ((const float4*)src)[t];
}

__global__ __launch_bounds__(256) void zero_kernel(float* __restrict__ p, int n) {
    int t = blockIdx.x * blockDim.x + threadIdx.x;
    if (t < n) p[t] = 0.0f;
}

// ---------------------------------------------------------------------------
// Edge scatter-add: Hb[dst] += X[src]   (Hb pre-initialized to X => h = x + agg)
// One warp-lane handles 4 features of one edge (float4 load, 4 f32 atomics).
// ---------------------------------------------------------------------------
__global__ __launch_bounds__(256) void scatter_add_kernel(const float* __restrict__ X,
                                                          const int* __restrict__ ei,
                                                          float* __restrict__ Hb) {
    int t = blockIdx.x * blockDim.x + threadIdx.x;
    int e = t >> 5;
    if (e >= EE) return;
    int f = (t & 31) << 2;
    int s = ei[e];
    int d = ei[EE + e];
    float4 v = *(const float4*)(X + (size_t)s * 128 + f);
    float* p = Hb + (size_t)d * 128 + f;
    atomicAdd(p + 0, v.x);
    atomicAdd(p + 1, v.y);
    atomicAdd(p + 2, v.z);
    atomicAdd(p + 3, v.w);
}

// ---------------------------------------------------------------------------
// Segment (graph) pooling: Gc[batch[n], colOff + f] += H[n, f]
// ---------------------------------------------------------------------------
__global__ __launch_bounds__(256) void pool_kernel(const float* __restrict__ H,
                                                   const int* __restrict__ batch,
                                                   float* __restrict__ Gc, int colOff) {
    int t = blockIdx.x * blockDim.x + threadIdx.x;
    int node = t >> 5;
    if (node >= NN) return;
    int f = (t & 31) << 2;
    int b = batch[node];
    float4 v = *(const float4*)(H + (size_t)node * 128 + f);
    float* p = Gc + (size_t)b * 384 + colOff + f;
    atomicAdd(p + 0, v.x);
    atomicAdd(p + 1, v.y);
    atomicAdd(p + 2, v.z);
    atomicAdd(p + 3, v.w);
}

// ---------------------------------------------------------------------------
// WMMA GEMM:  Out[rows, ncols] = epilogue( A[rows,K] @ W[K,ncols] )
//   mode 0: bias + BatchNorm(eval) + ReLU
//   mode 1: bias + ReLU
//   mode 2: bias only
// Block: 256 threads = 8 waves. Each wave: one 16-row M tile x 128 cols
// (8 n-tiles of 16). Block covers 128 rows x 128 cols (grid.y = ncols/128).
// W is staged transposed into LDS as f16 (Bs[col][k], stride 132 halfs).
// A fragment layout per CDNA5 ISA 16-bit A: lane L -> M=L%16,
//   halfs[0..7] = K g+0..7, halfs[8..15] = K g+16..23, g = (L/16)*8.
// B fragment: lane L -> N=L%16, same K pattern (column of W).
// D layout: lane L -> N=L%16, acc[r] -> M = g + r.
// ---------------------------------------------------------------------------
__global__ __launch_bounds__(256) void gemm_wmma_kernel(
    const float* __restrict__ A,
    const float* __restrict__ W,
    const float* __restrict__ bias,
    const float* __restrict__ gamma,
    const float* __restrict__ beta,
    const float* __restrict__ mean,
    const float* __restrict__ var,
    float* __restrict__ Out,
    int rows, int K, int ncols, int mode)
{
    __shared__ _Float16 Bs[128 * 132];   // 33 KB, W sub-panel transposed, f16

    const int tid  = threadIdx.x;
    const int lane = tid & 31;
    const int wave = tid >> 5;
    const int m    = lane & 15;
    const int g    = (lane >> 4) * 8;
    const int rowBase = blockIdx.x * 128 + wave * 16;
    const int colBase = blockIdx.y * 128;
    int arow = rowBase + m;
    if (arow > rows - 1) arow = rows - 1;     // clamp for safe loads

    v8f acc[8];
#pragma unroll
    for (int i = 0; i < 8; i++) acc[i] = (v8f){0.f,0.f,0.f,0.f,0.f,0.f,0.f,0.f};

    for (int k0 = 0; k0 < K; k0 += 128) {
        __syncthreads();
        // Stage 128x128 panel of W, transposed + converted to f16.
        for (int i = tid; i < 128 * 128; i += 256) {
            int c  = i & 127;        // column (contiguous across threads -> coalesced)
            int kk = i >> 7;
            Bs[c * 132 + kk] = (_Float16)W[(size_t)(k0 + kk) * ncols + colBase + c];
        }
        __syncthreads();

#pragma unroll
        for (int ks = 0; ks < 4; ks++) {
            // ---- A fragment (global f32 -> f16) ----
            v16h a;
            const float* ap = A + (size_t)arow * K + k0 + ks * 32 + g;
#pragma unroll
            for (int e = 0; e < 8; e++) a[e]     = (_Float16)ap[e];
#pragma unroll
            for (int e = 0; e < 8; e++) a[8 + e] = (_Float16)ap[16 + e];

#pragma unroll
            for (int nt = 0; nt < 8; nt++) {
                // ---- B fragment from LDS ----
                v16h b;
                const _Float16* bp = &Bs[(nt * 16 + m) * 132 + ks * 32 + g];
#pragma unroll
                for (int e = 0; e < 8; e++) b[e]     = bp[e];
#pragma unroll
                for (int e = 0; e < 8; e++) b[8 + e] = bp[16 + e];

                acc[nt] = __builtin_amdgcn_wmma_f32_16x16x32_f16(
                    false, a, false, b, (short)0, acc[nt], false, false);
            }
        }
    }

    // ---- Epilogue ----
#pragma unroll
    for (int nt = 0; nt < 8; nt++) {
        int c = colBase + nt * 16 + m;
        float bv = bias[c];
        float s = 1.0f, t = 0.0f;
        if (mode == 0) {
            float sc = gamma[c] * rsqrtf(var[c] + BN_EPS);
            s = sc;
            t = beta[c] - mean[c] * sc;
        }
#pragma unroll
        for (int r = 0; r < 8; r++) {
            int orow = rowBase + g + r;
            if (orow < rows) {
                float v = acc[nt][r] + bv;
                v = v * s + t;
                if (mode <= 1) v = fmaxf(v, 0.0f);
                Out[(size_t)orow * ncols + c] = v;
            }
        }
    }
}

// ---------------------------------------------------------------------------
// Final projection: out[r] = M2[r,:] . W4 + b4   (256 x 256 -> 256)
// One wave per row, shuffle reduction.
// ---------------------------------------------------------------------------
__global__ __launch_bounds__(32) void lin4_kernel(const float* __restrict__ M2,
                                                  const float* __restrict__ W4,
                                                  const float* __restrict__ b4,
                                                  float* __restrict__ out) {
    int r = blockIdx.x;
    int lane = threadIdx.x;
    float v = 0.0f;
    for (int j = lane; j < 256; j += 32)
        v += M2[(size_t)r * 256 + j] * W4[j];
    for (int off = 16; off > 0; off >>= 1)
        v += __shfl_xor(v, off, 32);
    if (lane == 0) out[r] = v + b4[0];
}

// ---------------------------------------------------------------------------
// Host orchestration
// ---------------------------------------------------------------------------
extern "C" void kernel_launch(void* const* d_in, const int* in_sizes, int n_in,
                              void* d_out, int out_size, void* d_ws, size_t ws_size,
                              hipStream_t stream) {
    const float* x     = (const float*)d_in[0];
    const int*   ei    = (const int*)d_in[1];
    const int*   batch = (const int*)d_in[2];

    // Workspace layout (floats)
    float* ws   = (float*)d_ws;
    float* hbuf = ws;                              // [N,128]
    float* tbuf = hbuf + (size_t)NN * 128;         // [N,128]
    float* hcur = tbuf + (size_t)NN * 128;         // [N,128]
    float* gcat = hcur + (size_t)NN * 128;         // [256,384]
    float* m1   = gcat + (size_t)GG * 384;         // [256,256]
    float* m2   = m1   + (size_t)GG * 256;         // [256,256]

    const int n4     = NN * 32;                    // float4 count for [N,128]
    const int gemmBx = (NN + 127) / 128;           // 391

    // zero pooled-feature accumulator
    {
        int n = GG * 384;
        zero_kernel<<<(n + 255) / 256, 256, 0, stream>>>(gcat, n);
    }

    const float* cur = x;
    for (int layer = 0; layer < 3; layer++) {
        const float* W1    = (const float*)d_in[3 + layer * 8 + 0];
        const float* b1    = (const float*)d_in[3 + layer * 8 + 1];
        const float* gam   = (const float*)d_in[3 + layer * 8 + 2];
        const float* bet   = (const float*)d_in[3 + layer * 8 + 3];
        const float* mn    = (const float*)d_in[3 + layer * 8 + 4];
        const float* vr    = (const float*)d_in[3 + layer * 8 + 5];
        const float* W2    = (const float*)d_in[3 + layer * 8 + 6];
        const float* b2    = (const float*)d_in[3 + layer * 8 + 7];

        // h = cur + segment_sum(cur[src] -> dst)
        copy4_kernel<<<(n4 + 255) / 256, 256, 0, stream>>>(cur, hbuf, n4);
        scatter_add_kernel<<<(EE * 32 + 255) / 256, 256, 0, stream>>>(cur, ei, hbuf);

        // t = relu(BN(h @ W1 + b1))
        gemm_wmma_kernel<<<dim3(gemmBx, 1), 256, 0, stream>>>(
            hbuf, W1, b1, gam, bet, mn, vr, tbuf, NN, 128, 128, 0);

        // hcur = relu(t @ W2 + b2)
        gemm_wmma_kernel<<<dim3(gemmBx, 1), 256, 0, stream>>>(
            tbuf, W2, b2, nullptr, nullptr, nullptr, nullptr, hcur, NN, 128, 128, 1);

        // pool layer output into concatenated graph features
        pool_kernel<<<(NN * 32 + 255) / 256, 256, 0, stream>>>(hcur, batch, gcat, layer * 128);

        cur = hcur;
    }

    const float* lin1W = (const float*)d_in[27];
    const float* lin1b = (const float*)d_in[28];
    const float* lin2W = (const float*)d_in[29];
    const float* lin2b = (const float*)d_in[30];
    const float* lin4W = (const float*)d_in[31];
    const float* lin4b = (const float*)d_in[32];

    // m1 = relu(gcat[256,384] @ lin1W[384,256] + b)
    gemm_wmma_kernel<<<dim3(2, 2), 256, 0, stream>>>(
        gcat, lin1W, lin1b, nullptr, nullptr, nullptr, nullptr, m1, GG, 384, 256, 1);

    // m2 = relu(m1[256,256] @ lin2W[256,256] + b)
    gemm_wmma_kernel<<<dim3(2, 2), 256, 0, stream>>>(
        m1, lin2W, lin2b, nullptr, nullptr, nullptr, nullptr, m2, GG, 256, 256, 1);

    // out = m2 @ lin4W + b
    lin4_kernel<<<GG, 32, 0, stream>>>(m2, lin4W, lin4b, (float*)d_out);
}
